// SelfAttention_9912784519238
// MI455X (gfx1250) — compile-verified, matrix operand order checked by reading, not run on hardware
//
#include <hip/hip_runtime.h>

// Linformer attention for MI455X (gfx1250, wave32, WMMA 16x16x32 bf16).
// Pipeline (all matmul operands bf16, f32 accumulate):
//   k_cvt  : fp32 -> bf16 for Wq/Wk/Wv/We/Wf (packed v_cvt_pk_bf16_f32)
//   k_qkv  : Q = I@Wq^T (row-major), Kt = (I@Wk^T)^T, Vt = (I@Wv^T)^T
//   k_proj : Tk = We@K  [k][d] row-major,  Vft = (Wf@V)^T [d][k]
//   k_attn : PBar = Q@Tk^T * 1/16, softmax over k in-register,
//            LDS restage -> Head = P@Vf, fp32 out. PBar never hits HBM.

typedef __attribute__((ext_vector_type(16))) __bf16 v16bf;
typedef __attribute__((ext_vector_type(2)))  __bf16 v2bf;
typedef __attribute__((ext_vector_type(8)))  float  v8f;

#define BB 4
#define NN 8192
#define DD 256
#define KD 256   // projected sequence length k

#if __has_builtin(__builtin_amdgcn_cvt_pk_bf16_f32)
#define USE_CVT_PK 1
#endif

// Pack two f32 -> two bf16 in one dword (v_cvt_pk_bf16_f32 when available).
__device__ __forceinline__ unsigned int f2bf_pk(float lo, float hi) {
#ifdef USE_CVT_PK
  union { v2bf v; unsigned int u; } t;
  t.v = __builtin_amdgcn_cvt_pk_bf16_f32(lo, hi);
  return t.u;
#else
  unsigned int a = __float_as_uint(lo);
  a += 0x7FFFu + ((a >> 16) & 1u);
  unsigned int b = __float_as_uint(hi);
  b += 0x7FFFu + ((b >> 16) & 1u);
  return (a >> 16) | (b & 0xFFFF0000u);
#endif
}
__device__ __forceinline__ unsigned short f2bf(float x) {
  return (unsigned short)f2bf_pk(x, x);
}

union FragU { v16bf v; uint4 u[2]; };

// A-style fragment (16x32 bf16) from row-major memory with leading dim ld.
// Lane L (0-15): row L, K 0..7 & 16..23 ; lane L+16: row L, K 8..15 & 24..31.
// B fragments are loaded with the same routine from B^T stored row-major.
__device__ __forceinline__ v16bf load_frag(const unsigned short* __restrict__ base, int ld) {
  const int lane = threadIdx.x & 31;
  const int r = lane & 15, hi = lane >> 4;
  const unsigned short* p = base + (size_t)r * ld + hi * 8;
  FragU f;
  f.u[0] = *reinterpret_cast<const uint4*>(p);
  f.u[1] = *reinterpret_cast<const uint4*>(p + 16);
  return f.v;
}

// Same fragment, sourced from fp32 memory with inline packed f32->bf16.
__device__ __forceinline__ v16bf load_frag_f32(const float* __restrict__ base, int ld) {
  const int lane = threadIdx.x & 31;
  const int r = lane & 15, hi = lane >> 4;
  const float* p = base + (size_t)r * ld + hi * 8;
  float4 a0 = *reinterpret_cast<const float4*>(p + 0);
  float4 a1 = *reinterpret_cast<const float4*>(p + 4);
  float4 a2 = *reinterpret_cast<const float4*>(p + 16);
  float4 a3 = *reinterpret_cast<const float4*>(p + 20);
  float t[16] = {a0.x, a0.y, a0.z, a0.w, a1.x, a1.y, a1.z, a1.w,
                 a2.x, a2.y, a2.z, a2.w, a3.x, a3.y, a3.z, a3.w};
  union { v16bf v; unsigned int u[8]; } f;
#pragma unroll
  for (int i = 0; i < 8; ++i) f.u[i] = f2bf_pk(t[2 * i], t[2 * i + 1]);
  return f.v;
}

__device__ __forceinline__ v8f wmma_bf16(v16bf a, v16bf b, v8f c) {
  return __builtin_amdgcn_wmma_f32_16x16x32_bf16(false, a, false, b, (short)0, c,
                                                 false, false);
}

// Transposed bf16 tile store: D-layout lane holds column c over rows 0..7(+8hi);
// in the transposed target those 8 rows are 8 CONSECUTIVE ushorts -> pack all
// 8 values into one uint4 and issue a single global_store_b128 per lane.
// dst = tile origin in T[col][row] layout, ld = leading dim of T.
__device__ __forceinline__ void store_tileT_bf16(unsigned short* __restrict__ dst,
                                                 int ld, v8f a) {
  const int lane = threadIdx.x & 31;
  const int c = lane & 15, hi = lane >> 4;
  union { uint4 q; unsigned int w[4]; } t;
#pragma unroll
  for (int j = 0; j < 4; ++j) t.w[j] = f2bf_pk(a[2 * j], a[2 * j + 1]);
  *reinterpret_cast<uint4*>(dst + (size_t)c * ld + 8 * hi) = t.q;
}

// ---------------------------------------------------------------- conversions
__global__ void k_cvt(const float* __restrict__ in, unsigned int* __restrict__ out,
                      int n2) {  // n2 = element count / 2
  int i = blockIdx.x * blockDim.x + threadIdx.x;
  if (i < n2) {
    float2 v = reinterpret_cast<const float2*>(in)[i];
    out[i] = f2bf_pk(v.x, v.y);
  }
}

// ---------------------------------------------------------------- Q/K/V GEMMs
// One wave per 16-row block of one batch. A-fragments of I are loaded once
// (fp32->bf16 inline) and reused across all 3 weights x 16 col tiles.
__global__ __launch_bounds__(256) void k_qkv(
    const float* __restrict__ I,
    const unsigned short* __restrict__ Wq, const unsigned short* __restrict__ Wk,
    const unsigned short* __restrict__ Wv,
    unsigned short* __restrict__ Q,   // [B][N][D] row-major
    unsigned short* __restrict__ Kt,  // [B][D][N] (K transposed)
    unsigned short* __restrict__ Vt)  // [B][D][N] (V transposed)
{
  const int wid = blockIdx.x * 8 + (threadIdx.x >> 5);  // 2048 waves
  const int b  = wid >> 9;
  const int n0 = (wid & 511) << 4;
  const int lane = threadIdx.x & 31;
  const int c = lane & 15, hi = lane >> 4;

  const float* Ib = I + (size_t)b * NN * DD;
  v16bf aI[8];
#pragma unroll
  for (int dt = 0; dt < 8; ++dt)
    aI[dt] = load_frag_f32(Ib + (size_t)n0 * DD + dt * 32, DD);

  for (int mat = 0; mat < 3; ++mat) {
    const unsigned short* W = (mat == 0) ? Wq : (mat == 1 ? Wk : Wv);
    for (int et = 0; et < 16; ++et) {
      v8f acc = {};
#pragma unroll
      for (int dt = 0; dt < 8; ++dt) {
        v16bf bf = load_frag(W + (size_t)(et * 16) * DD + dt * 32, DD);
        acc = wmma_bf16(aI[dt], bf, acc);
      }
      if (mat == 0) {
        unsigned short* dst = Q + ((size_t)b * NN + n0) * DD + et * 16;
#pragma unroll
        for (int r = 0; r < 8; ++r)
          dst[(size_t)(r + 8 * hi) * DD + c] = f2bf(acc[r]);
      } else {
        unsigned short* T = ((mat == 1) ? Kt : Vt) + (size_t)b * DD * NN;
        store_tileT_bf16(T + (size_t)(et * 16) * NN + n0, NN, acc);
      }
    }
  }
}

// --------------------------------------------------- sequence projections
// Tk[b][k][d] = sum_n We[k][n]*K[b][n][d]  (A = We rows, B^T = Kt rows)
// Vft[b][d][k] = sum_n Wf[k][n]*V[b][n][d] stored transposed.
// 1024 waves: (b, mat, k-tile, d-group of 2 tiles).
__global__ __launch_bounds__(256) void k_proj(
    const unsigned short* __restrict__ We, const unsigned short* __restrict__ Wf,
    const unsigned short* __restrict__ Kt, const unsigned short* __restrict__ Vt,
    unsigned short* __restrict__ Tk,   // [B][K][D]
    unsigned short* __restrict__ Vft)  // [B][D][K]
{
  const int wid = blockIdx.x * 8 + (threadIdx.x >> 5);  // 1024 waves
  const int b   = wid >> 8;
  const int mat = (wid >> 7) & 1;
  const int k0  = ((wid >> 3) & 15) * 16;
  const int dg  = wid & 7;
  const int lane = threadIdx.x & 31;
  const int c = lane & 15, hi = lane >> 4;

  const unsigned short* A  = mat ? Wf : We;                          // [K][N]
  const unsigned short* Bt = (mat ? Vt : Kt) + (size_t)b * DD * NN;  // [D][N]

  v8f acc[2] = {{}, {}};
  for (int ns = 0; ns < NN / 32; ++ns) {
    const int n0 = ns * 32;
    if (ns + 1 < NN / 32)   // gfx1250 global_prefetch_b8 on next A stripe
      __builtin_prefetch(A + (size_t)k0 * NN + n0 + 32, 0, 1);
    v16bf aW = load_frag(A + (size_t)k0 * NN + n0, NN);
#pragma unroll
    for (int j = 0; j < 2; ++j) {
      const int d0 = dg * 32 + j * 16;
      v16bf bf = load_frag(Bt + (size_t)d0 * NN + n0, NN);
      acc[j] = wmma_bf16(aW, bf, acc[j]);
    }
  }
#pragma unroll
  for (int j = 0; j < 2; ++j) {
    const int d0 = dg * 32 + j * 16;
    if (mat == 0) {
      unsigned short* dst = Tk + ((size_t)b * KD + k0) * DD + d0;
#pragma unroll
      for (int r = 0; r < 8; ++r)
        dst[(size_t)(r + 8 * hi) * DD + c] = f2bf(acc[j][r]);
    } else {
      store_tileT_bf16(Vft + ((size_t)b * DD + d0) * KD + k0, KD, acc[j]);
    }
  }
}

// ------------------------------------------- fused PBar + softmax + Head
// One wave owns 16 query rows: full 16x256 PBar block lives in 16 f32
// accumulators. C/D layout: lane = column (within half-wave), VGPR = row,
// so softmax is a register reduction over 16 tiles + 4-step shfl_xor
// butterfly over lanes 0..15 / 16..31. Then an LDS restage converts the
// D-layout block into A-fragments for the Head GEMM.
__global__ __launch_bounds__(128) void k_attn(
    const unsigned short* __restrict__ Q,    // [B][N][D]
    const unsigned short* __restrict__ Tk,   // [B][K][D]
    const unsigned short* __restrict__ Vft,  // [B][D][K]
    float* __restrict__ Out)                 // [B][N][D] fp32
{
  __shared__ unsigned short smem[4][16][KD];  // 32 KB, wave-private slices
  const int wslot = threadIdx.x >> 5;
  const int wid = blockIdx.x * 4 + wslot;     // 2048 waves
  const int b  = wid >> 9;
  const int n0 = (wid & 511) << 4;
  const int lane = threadIdx.x & 31;
  const int c = lane & 15, hi = lane >> 4;

  const unsigned short* Qb  = Q   + (size_t)b * NN * DD;
  const unsigned short* Tkb = Tk  + (size_t)b * KD * DD;
  const unsigned short* Vb  = Vft + (size_t)b * DD * KD;

  v16bf aQ[8];
#pragma unroll
  for (int dt = 0; dt < 8; ++dt)
    aQ[dt] = load_frag(Qb + (size_t)n0 * DD + dt * 32, DD);

  v8f acc[16];
#pragma unroll
  for (int kt = 0; kt < 16; ++kt) {
    v8f a = {};
#pragma unroll
    for (int dt = 0; dt < 8; ++dt) {
      v16bf bf = load_frag(Tkb + (size_t)(kt * 16) * DD + dt * 32, DD);
      a = wmma_bf16(aQ[dt], bf, a);
    }
    acc[kt] = a;
  }

  const float scale = 0.0625f;  // 1/sqrt(256)
#pragma unroll
  for (int r = 0; r < 8; ++r) {
    float mx = acc[0][r];
#pragma unroll
    for (int kt = 1; kt < 16; ++kt) mx = fmaxf(mx, acc[kt][r]);
#pragma unroll
    for (int off = 8; off > 0; off >>= 1)
      mx = fmaxf(mx, __shfl_xor(mx, off, 32));
    float sum = 0.f;
#pragma unroll
    for (int kt = 0; kt < 16; ++kt) {
      float e = __expf(scale * (acc[kt][r] - mx));
      acc[kt][r] = e;
      sum += e;
    }
#pragma unroll
    for (int off = 8; off > 0; off >>= 1)
      sum += __shfl_xor(sum, off, 32);
    float inv = 1.0f / sum;
#pragma unroll
    for (int kt = 0; kt < 16; ++kt) acc[kt][r] *= inv;
  }

  // restage P (D-layout) -> LDS row-major [16][256]
#pragma unroll
  for (int kt = 0; kt < 16; ++kt)
#pragma unroll
    for (int r = 0; r < 8; ++r)
      smem[wslot][r + 8 * hi][kt * 16 + c] = f2bf(acc[kt][r]);
  __syncthreads();

  v16bf aP[8];
#pragma unroll
  for (int kk = 0; kk < 8; ++kk)
    aP[kk] = load_frag(&smem[wslot][0][0] + kk * 32, KD);

  for (int dt = 0; dt < 16; ++dt) {
    v8f o = {};
#pragma unroll
    for (int kk = 0; kk < 8; ++kk) {
      v16bf bf = load_frag(Vb + (size_t)(dt * 16) * KD + kk * 32, KD);
      o = wmma_bf16(aP[kk], bf, o);
    }
    float* dst = Out + ((size_t)b * NN + n0) * DD + dt * 16;
#pragma unroll
    for (int r = 0; r < 8; ++r)
      dst[(size_t)(r + 8 * hi) * DD + c] = o[r];
  }
}

// -------------------------------------------------------------------- launch
extern "C" void kernel_launch(void* const* d_in, const int* in_sizes, int n_in,
                              void* d_out, int out_size, void* d_ws, size_t ws_size,
                              hipStream_t stream) {
  const float* I  = (const float*)d_in[0];
  const float* Wq = (const float*)d_in[1];
  const float* Wk = (const float*)d_in[2];
  const float* Wv = (const float*)d_in[3];
  const float* We = (const float*)d_in[4];
  const float* Wf = (const float*)d_in[5];
  float* Out = (float*)d_out;

  char* ws = (char*)d_ws;
  size_t off = 0;
  auto alloc = [&](size_t bytes) -> unsigned short* {
    unsigned short* p = (unsigned short*)(ws + off);
    off += (bytes + 255) & ~(size_t)255;
    return p;
  };
  unsigned short* Wq_bf = alloc((size_t)DD * DD * 2);
  unsigned short* Wk_bf = alloc((size_t)DD * DD * 2);
  unsigned short* Wv_bf = alloc((size_t)DD * DD * 2);
  unsigned short* We_bf = alloc((size_t)KD * NN * 2);
  unsigned short* Wf_bf = alloc((size_t)KD * NN * 2);
  unsigned short* Qbf   = alloc((size_t)BB * NN * DD * 2);
  unsigned short* Ktbf  = alloc((size_t)BB * DD * NN * 2);
  unsigned short* Vtbf  = alloc((size_t)BB * DD * NN * 2);
  unsigned short* TkB   = alloc((size_t)BB * KD * DD * 2);
  unsigned short* VftB  = alloc((size_t)BB * DD * KD * 2);
  (void)ws_size; (void)in_sizes; (void)n_in; (void)out_size;

  const int nw2 = DD * DD / 2;      // 32768 dword pairs
  const int ns2 = KD * NN / 2;      // 1048576 dword pairs
  k_cvt<<<(nw2 + 255) / 256, 256, 0, stream>>>(Wq, (unsigned int*)Wq_bf, nw2);
  k_cvt<<<(nw2 + 255) / 256, 256, 0, stream>>>(Wk, (unsigned int*)Wk_bf, nw2);
  k_cvt<<<(nw2 + 255) / 256, 256, 0, stream>>>(Wv, (unsigned int*)Wv_bf, nw2);
  k_cvt<<<(ns2 + 255) / 256, 256, 0, stream>>>(We, (unsigned int*)We_bf, ns2);
  k_cvt<<<(ns2 + 255) / 256, 256, 0, stream>>>(Wf, (unsigned int*)Wf_bf, ns2);

  k_qkv <<<256, 256, 0, stream>>>(I, Wq_bf, Wk_bf, Wv_bf, Qbf, Ktbf, Vtbf);
  k_proj<<<128, 256, 0, stream>>>(We_bf, Wf_bf, Ktbf, Vtbf, TkB, VftB);
  k_attn<<<512, 128, 0, stream>>>(Qbf, TkB, VftB, Out);
}
